// DSLRCollisionDecoder_31903017075432
// MI455X (gfx1250) — matrix operand order repaired and checked by reading, not compiled, not run on hardware
//
#include <hip/hip_runtime.h>
#include <math.h>

// ---- CDNA5 WMMA vector types ----
typedef _Float16 v8h  __attribute__((ext_vector_type(8)));
typedef _Float16 v16h __attribute__((ext_vector_type(16)));
typedef float    v8f  __attribute__((ext_vector_type(8)));

union V16 { v16h v; v8h h[2]; };

#define BB   256
#define NN   64
#define KK   32
#define ROWS 128     // pairs handled per block (8 blocks per batch)
#define OUTC 102

__device__ __forceinline__ float gelu_exact(float x) {
    return 0.5f * x * (1.0f + erff(x * 0.70710678118654752440f));
}

__global__ __launch_bounds__(256) void fused_decoder(
    const float* __restrict__ z_a,   const float* __restrict__ z_b,
    const float* __restrict__ fps_a, const float* __restrict__ fps_b,
    const int*   __restrict__ a_idx, const int*   __restrict__ b_idx,
    const float* __restrict__ pos_w1, const float* __restrict__ pos_b1,
    const float* __restrict__ pos_w2, const float* __restrict__ pos_b2,
    const float* __restrict__ pw_w1,  const float* __restrict__ pw_b1,
    const float* __restrict__ pw_w2,  const float* __restrict__ pw_b2,
    const float* __restrict__ pw_w3,  const float* __restrict__ pw_b3,
    float* __restrict__ out)
{
    // Weights transposed to [n][k], K padded to 64 -> B-fragment = 1 contiguous v16h load.
    __shared__ __align__(32) _Float16 sW[3][64][64];
    __shared__ float sBias[3][64];
    __shared__ float sPw1[16], sPb1[16], sPw2[256], sPb2[16];
    __shared__ __align__(32) _Float16 sFeat[ROWS][64];  // feat, later reused as x2
    __shared__ __align__(32) _Float16 sX1[ROWS][64];    // x1 (f16 GEMM input for layer 2)

    const int tid      = threadIdx.x;
    const int b        = blockIdx.x >> 3;
    const int pairBase = (blockIdx.x & 7) * ROWS;

    // ---------------- Phase A: stage weights ----------------
    for (int e = tid; e < 64 * 64; e += 256) {
        const int n = e >> 6, k = e & 63;
        sW[0][n][k] = (k < 48) ? (_Float16)pw_w1[k * 64 + n] : (_Float16)0.0f;
        sW[1][n][k] = (_Float16)pw_w2[k * 64 + n];
        sW[2][n][k] = (_Float16)pw_w3[k * 64 + n];
    }
    if (tid < 64) {
        sBias[0][tid] = pw_b1[tid];
        sBias[1][tid] = pw_b2[tid];
        sBias[2][tid] = pw_b3[tid];
    }
    if (tid < 16) { sPw1[tid] = pos_w1[tid]; sPb1[tid] = pos_b1[tid]; sPb2[tid] = pos_b2[tid]; }
    sPw2[tid] = pos_w2[tid];   // blockDim == 256
    __syncthreads();

    // ---------------- Phase B: per-pair scalar features ----------------
    // Two threads cooperate on each pair: half 0 -> pz cols 0..3, pos outputs 0..7,
    // output channels [0,22); half 1 -> pz cols 4..7, pos outputs 8..15,
    // output channels [22,38) + zero-pad of feat. Frame math is duplicated (cheap).
    {
        const int r    = tid & 127;           // pair row within block
        const int half = tid >> 7;            // 0 or 1
        const int q    = pairBase + r;        // pair index within batch [0,1024)
        const int i    = q >> 5, j = q & 31;
        const int ia   = a_idx[b * KK + i];
        const int jb   = b_idx[b * KK + j];

        float fga[3], fgb[3];
        #pragma unroll
        for (int e = 0; e < 3; ++e) {
            fga[e] = fps_a[((size_t)b * NN + ia) * 3 + e];
            fgb[e] = fps_b[((size_t)b * NN + jb) * 3 + e];
        }

        float za[16], zb[16], sa = 0.f, sb = 0.f;
        const float* pa = z_a + ((size_t)b * NN + ia) * 16;
        const float* pb = z_b + ((size_t)b * NN + jb) * 16;
        #pragma unroll
        for (int e = 0; e < 16; ++e) {
            za[e] = pa[e]; zb[e] = pb[e];
            sa += za[e] * za[e]; sb += zb[e] * zb[e];
        }
        const float zna = sqrtf(sa), znb = sqrtf(sb);

        float pd0 = fga[0] - fgb[0], pd1 = fga[1] - fgb[1], pd2 = fga[2] - fgb[2];
        const float dist   = sqrtf(pd0 * pd0 + pd1 * pd1 + pd2 * pd2);
        const float z_norm = fmaxf(zna, znb);
        const float twod   = 2.0f * dist;
        const float scale  = (z_norm > twod) ? z_norm : twod;
        const bool  swap   = zna < znb;
        if (swap) { pd0 = -pd0; pd1 = -pd1; pd2 = -pd2; }
        // half 0 writes columns 0..3 (first source), half 1 columns 4..7 (second source)
        const float* src = (half == 0) ? (swap ? zb : za) : (swap ? za : zb);

        // R_inv rows = x, y, z  (z = normalized pd, x = gram-schmidt of [1,0,0], y = z×x)
        const float izn = 1.0f / (dist + 1e-8f);
        const float zx = pd0 * izn, zy = pd1 * izn, zz = pd2 * izn;
        float xx = 1.0f - zx * zx, xy = -zx * zy, xz = -zx * zz;
        const float xn  = sqrtf(xx * xx + xy * xy + xz * xz);
        const float ixn = 1.0f / (xn + 1e-8f);
        xx *= ixn; xy *= ixn; xz *= ixn;
        const float yx = zy * xz - zz * xy;
        const float yy = zz * xx - zx * xz;
        const float yz = zx * xy - zy * xx;
        const float invs = 1.0f / scale;

        _Float16* frow = &sFeat[r][0];
        #pragma unroll
        for (int cc = 0; cc < 4; ++cc) {
            const int c = 4 * half + cc;
            const float p0 = src[0 * 4 + cc], p1 = src[1 * 4 + cc];
            const float p2 = src[2 * 4 + cc], p3 = src[3 * 4 + cc];
            frow[0 * 8 + c] = (_Float16)(p0 * invs);
            frow[1 * 8 + c] = (_Float16)((xx * p1 + xy * p2 + xz * p3) * invs);
            frow[2 * 8 + c] = (_Float16)((yx * p1 + yy * p2 + yz * p3) * invs);
            frow[3 * 8 + c] = (_Float16)((zx * p1 + zy * p2 + zz * p3) * invs);
        }

        // positional MLP on rotated+scaled z-component of pd; h1 duplicated per half,
        // each half produces 8 of the 16 second-layer outputs.
        const float pdz = (zx * pd0 + zy * pd1 + zz * pd2) * invs;
        float h1[16];
        #pragma unroll
        for (int o = 0; o < 16; ++o) h1[o] = gelu_exact(pdz * sPw1[o] + sPb1[o]);
        #pragma unroll
        for (int oo = 0; oo < 8; ++oo) {
            const int o = 8 * half + oo;
            float s = sPb2[o];
            #pragma unroll
            for (int l = 0; l < 16; ++l) s += h1[l] * sPw2[l * 16 + o];
            frow[32 + o] = (_Float16)gelu_exact(s);
        }

        // direct gather/concat output channels + K-padding
        float* op = out + ((size_t)b * 1024 + q) * OUTC;
        if (half == 0) {
            op[0] = fga[0]; op[1] = fga[1]; op[2] = fga[2];
            op[3] = fgb[0]; op[4] = fgb[1]; op[5] = fgb[2];
            #pragma unroll
            for (int e = 0; e < 16; ++e) op[6 + e] = za[e];
        } else {
            #pragma unroll
            for (int e = 0; e < 16; ++e) op[22 + e] = zb[e];
            #pragma unroll
            for (int e = 48; e < 64; ++e) frow[e] = (_Float16)0.0f;
        }
    }
    __syncthreads();

    // ---------------- Phase C: WMMA MLP chain (wave-private 16-row tiles) ----------------
    const int lane = tid & 31;
    const int r0   = (tid >> 5) * 16;          // this wave's row tile
    const int arow = r0 + (lane & 15);         // A-matrix row for this lane
    const int ak   = lane & 16;                // A k chunk base: 0 or 16
    const int bn   = lane & 15;                // B column within n-tile
    const int bk   = lane & 16;                // B k base: 0 or 16
    const int dm   = r0 + ((lane & 16) >> 1);  // D row base (+v)

    auto runLayer = [&](const _Float16 (*Ain)[64], int l, v8f* acc) {
        V16 a0, a1;
        a0.h[0] = *(const v8h*)&Ain[arow][ak +  0];
        a0.h[1] = *(const v8h*)&Ain[arow][ak +  8];
        a1.h[0] = *(const v8h*)&Ain[arow][ak + 32];
        a1.h[1] = *(const v8h*)&Ain[arow][ak + 40];
        #pragma unroll
        for (int t = 0; t < 4; ++t) {
            const v16h b0 = *(const v16h*)&sW[l][16 * t + bn][bk];
            const v16h b1 = *(const v16h*)&sW[l][16 * t + bn][32 + bk];
            v8f c = {};
            c = __builtin_amdgcn_wmma_f32_16x16x32_f16(false, a0.v, false, b0,
                                                       (short)0, c, false, false);
            c = __builtin_amdgcn_wmma_f32_16x16x32_f16(false, a1.v, false, b1,
                                                       (short)0, c, false, false);
            acc[t] = c;
        }
    };

    v8f acc[4];
    v8f x1f[4];   // f32 residual kept in registers (reference adds f32 x1)

    // layer 1: feat(48 padded 64) -> 64, gelu; keep f32 in regs, store f16 copy for GEMM
    runLayer(sFeat, 0, acc);
    #pragma unroll
    for (int t = 0; t < 4; ++t)
        #pragma unroll
        for (int v = 0; v < 8; ++v) {
            const int m = dm + v, n = 16 * t + bn;
            const float g = gelu_exact(acc[t][v] + sBias[0][n]);
            x1f[t][v] = g;
            sX1[m][n] = (_Float16)g;
        }

    // layer 2: x1 -> 64, gelu, store x2 (reuse sFeat)
    runLayer(sX1, 1, acc);
    #pragma unroll
    for (int t = 0; t < 4; ++t)
        #pragma unroll
        for (int v = 0; v < 8; ++v) {
            const int m = dm + v, n = 16 * t + bn;
            sFeat[m][n] = (_Float16)gelu_exact(acc[t][v] + sBias[1][n]);
        }

    // layer 3: x2 -> 64, gelu, + f32 x1 residual, write embed channels [38,102)
    runLayer(sFeat, 2, acc);
    #pragma unroll
    for (int t = 0; t < 4; ++t)
        #pragma unroll
        for (int v = 0; v < 8; ++v) {
            const int m = dm + v, n = 16 * t + bn;
            const float e = gelu_exact(acc[t][v] + sBias[2][n]) + x1f[t][v];
            out[((size_t)b * 1024 + pairBase + m) * OUTC + 38 + n] = e;
        }
}

extern "C" void kernel_launch(void* const* d_in, const int* in_sizes, int n_in,
                              void* d_out, int out_size, void* d_ws, size_t ws_size,
                              hipStream_t stream) {
    (void)in_sizes; (void)n_in; (void)out_size; (void)d_ws; (void)ws_size;
    fused_decoder<<<dim3(BB * 8), dim3(256), 0, stream>>>(
        (const float*)d_in[0],  (const float*)d_in[1],
        (const float*)d_in[2],  (const float*)d_in[3],
        (const int*)  d_in[4],  (const int*)  d_in[5],
        (const float*)d_in[6],  (const float*)d_in[7],
        (const float*)d_in[8],  (const float*)d_in[9],
        (const float*)d_in[10], (const float*)d_in[11],
        (const float*)d_in[12], (const float*)d_in[13],
        (const float*)d_in[14], (const float*)d_in[15],
        (float*)d_out);
}
